// IConv_72138270703708
// MI455X (gfx1250) — matrix operand order repaired and checked by reading, not compiled, not run on hardware
//
#include <hip/hip_runtime.h>

// ---------------------------------------------------------------------------
// Dynamic per-sample conv filter, fused:
//   w[b,o,k,d] = relu(sum_m mhc[b,m,d] * weight[o,k,m])   (never hits HBM)
//   out[b,o,p] = sum_{k,d} w[b,o,k,d] * pep[b,p+k,d] + bias[o]
// B=4096, L=15, D=64, O=128, K=9, M=34, P=7.
// bf16 WMMA (v_wmma_f32_16x16x32_bf16) with f32 accumulation.
// ---------------------------------------------------------------------------

typedef __attribute__((ext_vector_type(16))) __bf16 v16bf;
typedef __attribute__((ext_vector_type(8)))  float  v8f;

#define B_TOT 4096
#define LPEP  15
#define DDIM  64
#define ODIM  128
#define KW    9
#define MH    34
#define MPAD  64          // m padded 34 -> 64 (two K=32 WMMA steps)
#define PW    7
#define PPAD  16
#define KD    (KW * DDIM) // 576

// LDS row strides (elements), padded to dodge 64-bank conflicts
#define MHCT_STRIDE 72
#define WIN_STRIDE  584
#define WPL_STRIDE  72

// Convert weight [O=128][K=9][M=34] f32 -> bf16 [k][o][m64], zero-padded m.
__global__ void prep_weight_kernel(const float* __restrict__ w,
                                   __bf16* __restrict__ wbf) {
    int idx = blockIdx.x * 256 + threadIdx.x;          // 9*128*64 = 73728
    if (idx >= KW * ODIM * MPAD) return;
    int m = idx & 63;
    int o = (idx >> 6) & 127;
    int k = idx >> 13;                                 // /(128*64)
    float v = (m < MH) ? w[(o * KW + k) * MH + m] : 0.0f;
    wbf[idx] = (__bf16)v;
}

__global__ __launch_bounds__(256)
void iconv_kernel(const float* __restrict__ pep,     // [B][15][64]
                  const float* __restrict__ mhc,     // [B][34][64]
                  const __bf16* __restrict__ wbf,    // [9][128][64] bf16
                  const float* __restrict__ bias,    // [128]
                  float* __restrict__ out)           // [B][128][7]
{
    __shared__ alignas(16) __bf16 mhcT[DDIM * MHCT_STRIDE];      // [d][m]   9.2 KB
    __shared__ alignas(16) __bf16 win [PPAD * WIN_STRIDE];       // [p][kd] 18.7 KB
    __shared__ alignas(16) __bf16 wpl [8 * 16 * WPL_STRIDE];     // per-wave 18.4 KB

    const int b    = blockIdx.x;
    const int tid  = threadIdx.x;
    const int lane = tid & 31;
    const int wave = tid >> 5;
    const int hl   = lane >> 4;   // half-wave: 0 or 1
    const int l16  = lane & 15;

    // --- cooperative fill: mhc^T (d-major, m zero-padded to 64), bf16 ---
    for (int idx = tid; idx < DDIM * MPAD; idx += 256) {          // 16 iters
        int m = idx & 63, d = idx >> 6;
        float v = (m < MH) ? mhc[((size_t)b * MH + m) * DDIM + d] : 0.0f;
        mhcT[d * MHCT_STRIDE + m] = (__bf16)v;
    }
    // --- cooperative fill: sliding windows win[p][k*64+d], p padded to 16 ---
    for (int idx = tid; idx < PPAD * KD; idx += 256) {            // 36 iters
        int p  = idx / KD;
        int kd = idx - p * KD;
        int k  = kd >> 6, d = kd & 63;
        float v = (p < PW) ? pep[((size_t)b * LPEP + p + k) * DDIM + d] : 0.0f;
        win[p * WIN_STRIDE + kd] = (__bf16)v;
    }
    __syncthreads();

    const int o0 = wave * 16;                    // this wave's o-block
    __bf16* plane = &wpl[wave * 16 * WPL_STRIDE];

    v8f acc2 = {};                               // out tile: 16 o x 16 p (f32)

    for (int k = 0; k < KW; ++k) {
        // ---- step 1: plane[o',d] = relu( W[k, o0+o', :] @ mhcT ), 16x64 ----
        for (int dt = 0; dt < 4; ++dt) {
            v8f c1 = {};
            #pragma unroll
            for (int ks = 0; ks < 2; ++ks) {     // m-chunks 0..31, 32..63
                // A fragment: rows o (M), cols m (K); two contiguous 16B loads
                const __bf16* arow = wbf +
                    ((size_t)(k * ODIM + o0 + l16)) * MPAD + ks * 32 + hl * 8;
                union { v16bf v; uint4 q[2]; } A;
                A.q[0] = *(const uint4*)(arow);
                A.q[1] = *(const uint4*)(arow + 16);
                // B fragment: rows m (K), cols d (N); two contiguous 16B loads
                const __bf16* brow =
                    &mhcT[(dt * 16 + l16) * MHCT_STRIDE + ks * 32 + hl * 8];
                union { v16bf v; uint4 q[2]; } Bf;
                Bf.q[0] = *(const uint4*)(brow);
                Bf.q[1] = *(const uint4*)(brow + 16);
                c1 = __builtin_amdgcn_wmma_f32_16x16x32_bf16(
                        false, A.v, false, Bf.v, (short)0, c1, false, false);
            }
            // relu + bf16, scatter into per-wave LDS plane (lane transpose)
            #pragma unroll
            for (int v = 0; v < 8; ++v) {
                int row = v + hl * 8;            // C layout: M = v + 8*(lane/16)
                float val = c1[v];
                val = val > 0.0f ? val : 0.0f;
                plane[row * WPL_STRIDE + dt * 16 + l16] = (__bf16)val;
            }
        }
        // intra-wave LDS store -> load ordering (CDNA5 split DS counter)
        asm volatile("s_wait_dscnt 0" ::: "memory");

        // ---- step 2: acc2 += plane(16 o x 64 d) @ win(64 d x 16 p) ----
        #pragma unroll
        for (int dh = 0; dh < 2; ++dh) {         // d-chunks 0..31, 32..63
            const __bf16* arow = &plane[l16 * WPL_STRIDE + dh * 32 + hl * 8];
            union { v16bf v; uint4 q[2]; } A;
            A.q[0] = *(const uint4*)(arow);
            A.q[1] = *(const uint4*)(arow + 16);
            const __bf16* brow =
                &win[l16 * WIN_STRIDE + k * DDIM + dh * 32 + hl * 8];
            union { v16bf v; uint4 q[2]; } Bf;
            Bf.q[0] = *(const uint4*)(brow);
            Bf.q[1] = *(const uint4*)(brow + 16);
            acc2 = __builtin_amdgcn_wmma_f32_16x16x32_bf16(
                      false, A.v, false, Bf.v, (short)0, acc2, false, false);
        }
    }

    // ---- epilogue: out[b][o0+row][p] for p = l16 < 7, + bias ----
    if (l16 < PW) {
        #pragma unroll
        for (int v = 0; v < 8; ++v) {
            int o = o0 + v + hl * 8;
            out[((size_t)b * ODIM + o) * PW + l16] = acc2[v] + bias[o];
        }
    }
}

extern "C" void kernel_launch(void* const* d_in, const int* in_sizes, int n_in,
                              void* d_out, int out_size, void* d_ws, size_t ws_size,
                              hipStream_t stream) {
    const float* pep  = (const float*)d_in[0];   // peptide_x [4096,15,64]
    const float* mhc  = (const float*)d_in[1];   // mhc_x     [4096,34,64]
    const float* wgt  = (const float*)d_in[2];   // weight    [128,9,34]
    const float* bias = (const float*)d_in[3];   // bias      [128]
    float* outp = (float*)d_out;                 // [4096,128,7]

    __bf16* wbf = (__bf16*)d_ws;                 // 147,456 bytes of scratch

    const int wtot = KW * ODIM * MPAD;           // 73728
    prep_weight_kernel<<<(wtot + 255) / 256, 256, 0, stream>>>(wgt, wbf);
    iconv_kernel<<<B_TOT, 256, 0, stream>>>(pep, mhc, wbf, bias, outp);
}
